// Model_3796751090166
// MI455X (gfx1250) — compile-verified, matrix operand order checked by reading, not run on hardware
//
#include <hip/hip_runtime.h>
#include <hip/hip_bf16.h>
#include <stdint.h>

typedef __bf16 bf16_t;
typedef __attribute__((ext_vector_type(16))) __bf16 v16bf;
typedef __attribute__((ext_vector_type(8)))  __bf16 v8bf;
typedef __attribute__((ext_vector_type(8)))  float  v8f;

// ---------------------------------------------------------------------------
// Elementwise / utility kernels
// ---------------------------------------------------------------------------

__global__ void k_zero_f32(float* __restrict__ p, long long n) {
  long long i = (long long)blockIdx.x * blockDim.x + threadIdx.x;
  long long stride = (long long)gridDim.x * blockDim.x;
  for (; i < n; i += stride) p[i] = 0.0f;
}

// Wt[n*K + k] = bf16(W[k*N + n])   (cast + transpose: W is [K,N] row-major)
__global__ void k_cast_transpose(const float* __restrict__ W, bf16_t* __restrict__ Wt,
                                 int K, int N) {
  long long total = (long long)K * N;
  long long i = (long long)blockIdx.x * blockDim.x + threadIdx.x;
  long long stride = (long long)gridDim.x * blockDim.x;
  for (; i < total; i += stride) {
    int k = (int)(i / N);
    int c = (int)(i % N);
    Wt[(long long)c * K + k] = (bf16_t)W[i];
  }
}

// agg[dst[e]][:] += x[src[e]][:]   (f32 atomics; agg array is L2-resident)
__global__ void k_scatter_add(const int* __restrict__ edges,   // [2,E] flat, int32
                              const float* __restrict__ x,     // [Nnodes, F]
                              float* __restrict__ agg,         // [Nnodes, F]
                              int E, int F) {
  int F4 = F >> 2;
  long long total = (long long)E * F4;
  long long i = (long long)blockIdx.x * blockDim.x + threadIdx.x;
  long long stride = (long long)gridDim.x * blockDim.x;
  for (; i < total; i += stride) {
    int e = (int)(i / F4);
    int c = ((int)(i % F4)) << 2;
    int s = edges[e];
    int d = edges[E + e];
    const float4 v = *reinterpret_cast<const float4*>(x + (long long)s * F + c);
    float* o = agg + (long long)d * F + c;
    atomicAdd(o + 0, v.x);
    atomicAdd(o + 1, v.y);
    atomicAdd(o + 2, v.z);
    atomicAdd(o + 3, v.w);
  }
}

// h = bf16((1+eps) * x + agg)
__global__ void k_combine_cast(const float* __restrict__ x, const float* __restrict__ agg,
                               const float* __restrict__ eps, bf16_t* __restrict__ h,
                               long long n) {
  float e1 = 1.0f + *eps;
  long long i = (long long)blockIdx.x * blockDim.x + threadIdx.x;
  long long stride = (long long)gridDim.x * blockDim.x;
  for (; i < n; i += stride) h[i] = (bf16_t)(fmaf(e1, x[i], agg[i]));
}

// ---------------------------------------------------------------------------
// WMMA bf16 GEMM:  C[M,N] = act(A[M,K] @ Bt^T + bias)
//   A  : [M,K] bf16 row-major
//   Bt : [N,K] bf16 row-major  (i.e. B transposed -> contiguous K per lane)
// Fragment layout (ISA 7.12.2, wave32):
//   lane = 16*half + ln;  A frag elems 0..7 -> k = kk + half*8 + 0..7,
//   elems 8..15 -> k = kk + 16 + half*8 + 0..7 ; B symmetric (column = ln).
//   C/D: vgpr r holds row (r + 8*half), col ln.
// K-loop is branch-free: partial N blocks clamp the sub-tile index (redundant
// compute on duplicates) and the boundary check lives only in the epilogue.
// ---------------------------------------------------------------------------

__device__ inline v16bf load_frag16(const bf16_t* __restrict__ p) {
  v8bf lo = *reinterpret_cast<const v8bf*>(p);        // 16B aligned
  v8bf hi = *reinterpret_cast<const v8bf*>(p + 16);   // 16B aligned
  v16bf f;
#pragma unroll
  for (int i = 0; i < 8; ++i) { f[i] = lo[i]; f[i + 8] = hi[i]; }
  return f;
}

__global__ __launch_bounds__(128)
void k_wmma_gemm(const bf16_t* __restrict__ A,
                 const bf16_t* __restrict__ Bt,
                 const float* __restrict__ bias,   // [N] or nullptr
                 float* __restrict__ Cf,           // f32 out or nullptr
                 bf16_t* __restrict__ Cb,          // bf16 out or nullptr
                 int M, int N, int K, int relu) {
  const int wave = threadIdx.x >> 5;
  const int lane = threadIdx.x & 31;
  const int half = lane >> 4;
  const int ln   = lane & 15;

  const int rowTile = blockIdx.y * 4 + wave;        // wave-uniform
  if (rowTile * 16 >= M) return;                    // uniform exit, EXEC stays full
  const int nTiles = N >> 4;
  const int ct0 = blockIdx.x * 4;

  // Clamp sub-tile indices so the K-loop needs no branches; duplicated
  // (clamped) tiles compute valid-but-redundant results and are not stored.
  const bf16_t* brow[4];
#pragma unroll
  for (int t = 0; t < 4; ++t) {
    int c = ct0 + t;
    if (c > nTiles - 1) c = nTiles - 1;
    brow[t] = Bt + (long long)(c * 16 + ln) * K + half * 8;
  }
  const bf16_t* arow = A + (long long)(rowTile * 16 + ln) * K + half * 8;

  v8f acc0 = {}, acc1 = {}, acc2 = {}, acc3 = {};
  for (int kk = 0; kk < K; kk += 32) {
    v16bf a = load_frag16(arow + kk);
    v16bf b0 = load_frag16(brow[0] + kk);
    v16bf b1 = load_frag16(brow[1] + kk);
    v16bf b2 = load_frag16(brow[2] + kk);
    v16bf b3 = load_frag16(brow[3] + kk);
    acc0 = __builtin_amdgcn_wmma_f32_16x16x32_bf16(false, a, false, b0, (short)0, acc0, false, false);
    acc1 = __builtin_amdgcn_wmma_f32_16x16x32_bf16(false, a, false, b1, (short)0, acc1, false, false);
    acc2 = __builtin_amdgcn_wmma_f32_16x16x32_bf16(false, a, false, b2, (short)0, acc2, false, false);
    acc3 = __builtin_amdgcn_wmma_f32_16x16x32_bf16(false, a, false, b3, (short)0, acc3, false, false);
  }

  v8f accv[4] = {acc0, acc1, acc2, acc3};
#pragma unroll
  for (int t = 0; t < 4; ++t) {
    int ct = ct0 + t;
    if (ct >= nTiles) continue;                     // uniform
    int col = ct * 16 + ln;
    float bs = bias ? bias[col] : 0.0f;
#pragma unroll
    for (int r = 0; r < 8; ++r) {
      float v = accv[t][r] + bs;
      if (relu) v = fmaxf(v, 0.0f);
      long long m = (long long)rowTile * 16 + r + half * 8;
      if (Cf) Cf[m * N + col] = v;
      if (Cb) Cb[m * N + col] = (bf16_t)v;
    }
  }
}

// ---------------------------------------------------------------------------
// Orchestration
// ---------------------------------------------------------------------------

extern "C" void kernel_launch(void* const* d_in, const int* in_sizes, int n_in,
                              void* d_out, int out_size, void* d_ws, size_t ws_size,
                              hipStream_t stream) {
  const int FG = 256, KE = 128;
  const int M = in_sizes[0] / FG;   // 10000
  const int D = in_sizes[1] / FG;   // 10000
  const int E = in_sizes[2] / 2;    // 320000
  const int NMAX = (M > D) ? M : D;

  const float* x_m = (const float*)d_in[0];
  const float* x_d = (const float*)d_in[1];
  const int*   e_mm = (const int*)d_in[2];
  const int*   e_dd = (const int*)d_in[3];
  auto F32 = [&](int i) { return (const float*)d_in[i]; };

  // ---- workspace carve (256B aligned slabs) ----
  char* ws = (char*)d_ws;
  size_t off = 0;
  auto alloc = [&](size_t bytes) -> void* {
    void* p = ws + off;
    off = (off + bytes + 255) & ~(size_t)255;
    return p;
  };

  auto castT = [&](int idx, int K_, int N_) -> bf16_t* {
    bf16_t* wt = (bf16_t*)alloc((size_t)K_ * N_ * sizeof(bf16_t));
    long long n = (long long)K_ * N_;
    k_cast_transpose<<<(int)((n + 255) / 256), 256, 0, stream>>>(F32(idx), wt, K_, N_);
    return wt;
  };

  // GIN + head weights -> bf16 [N,K] (transposed)
  bf16_t* wx1a = castT(4,  FG,     2 * FG);
  bf16_t* wx1b = castT(6,  2 * FG, 4 * FG);
  bf16_t* wx2a = castT(9,  4 * FG, 2 * FG);
  bf16_t* wx2b = castT(11, 2 * FG, FG);
  bf16_t* wy1a = castT(14, FG,     2 * FG);
  bf16_t* wy1b = castT(16, 2 * FG, 4 * FG);
  bf16_t* wy2a = castT(19, 4 * FG, 2 * FG);
  bf16_t* wy2b = castT(21, 2 * FG, FG);
  bf16_t* wlx1 = castT(24, FG, 256);
  bf16_t* wlx2 = castT(26, 256, 128);
  bf16_t* wlx3 = castT(28, 128, KE);
  bf16_t* wly1 = castT(30, FG, 256);
  bf16_t* wly2 = castT(32, 256, 128);
  bf16_t* wly3 = castT(34, 128, KE);

  // shared activation scratch (reused across both branches)
  float*  agg = (float*)alloc((size_t)NMAX * 1024 * sizeof(float));
  bf16_t* hb  = (bf16_t*)alloc((size_t)NMAX * 1024 * sizeof(bf16_t));
  bf16_t* tb  = (bf16_t*)alloc((size_t)NMAX * 512 * sizeof(bf16_t));
  float*  x1f = (float*)alloc((size_t)NMAX * 1024 * sizeof(float));
  bf16_t* xb  = (bf16_t*)alloc((size_t)NMAX * 256 * sizeof(bf16_t));
  bf16_t* h1  = (bf16_t*)alloc((size_t)NMAX * 256 * sizeof(bf16_t));
  bf16_t* h2  = (bf16_t*)alloc((size_t)NMAX * 128 * sizeof(bf16_t));
  bf16_t* xf  = (bf16_t*)alloc((size_t)M * 128 * sizeof(bf16_t));   // persistent
  bf16_t* yf  = (bf16_t*)alloc((size_t)D * 128 * sizeof(bf16_t));   // persistent

  auto zero = [&](float* p, long long n) {
    k_zero_f32<<<(int)((n + 255) / 256), 256, 0, stream>>>(p, n);
  };
  auto scatter = [&](const int* edges, const float* x, float* a, int F) {
    long long n = (long long)E * (F >> 2);
    k_scatter_add<<<(int)((n + 255) / 256), 256, 0, stream>>>(edges, x, a, E, F);
  };
  auto combine = [&](const float* x, const float* a, const float* eps, bf16_t* h, long long n) {
    k_combine_cast<<<(int)((n + 255) / 256), 256, 0, stream>>>(x, a, eps, h, n);
  };
  auto gemm = [&](const bf16_t* A, const bf16_t* Bt, const float* bias,
                  float* Cf, bf16_t* Cb, int M_, int N_, int K_, int relu) {
    dim3 grid((unsigned)((N_ / 16 + 3) / 4), (unsigned)((M_ / 16 + 3) / 4));
    k_wmma_gemm<<<grid, 128, 0, stream>>>(A, Bt, bias, Cf, Cb, M_, N_, K_, relu);
  };

  auto branch = [&](const float* X0, const int* edges, int NM,
                    bf16_t* w1a, const float* b1a, bf16_t* w1b, const float* b1b, const float* eps1,
                    bf16_t* w2a, const float* b2a, bf16_t* w2b, const float* b2b, const float* eps2,
                    bf16_t* wh1, const float* bh1, bf16_t* wh2, const float* bh2,
                    bf16_t* wh3, const float* bh3, bf16_t* outf) {
    // GIN layer 1: MLP((1+eps)x + segsum) -> relu -> X1 [NM, 4FG] f32
    zero(agg, (long long)NM * FG);
    scatter(edges, X0, agg, FG);
    combine(X0, agg, eps1, hb, (long long)NM * FG);
    gemm(hb, w1a, b1a, nullptr, tb, NM, 2 * FG, FG, 0);      // [NM,512] bf16
    gemm(tb, w1b, b1b, x1f, nullptr, NM, 4 * FG, 2 * FG, 1); // [NM,1024] f32, relu
    // GIN layer 2 -> relu -> X [NM, FG] bf16
    zero(agg, (long long)NM * 4 * FG);
    scatter(edges, x1f, agg, 4 * FG);
    combine(x1f, agg, eps2, hb, (long long)NM * 4 * FG);
    gemm(hb, w2a, b2a, nullptr, tb, NM, 2 * FG, 4 * FG, 0);
    gemm(tb, w2b, b2b, nullptr, xb, NM, FG, 2 * FG, 1);
    // dense heads (relu each)
    gemm(xb, wh1, bh1, nullptr, h1, NM, 256, FG, 1);
    gemm(h1, wh2, bh2, nullptr, h2, NM, 128, 256, 1);
    gemm(h2, wh3, bh3, nullptr, outf, NM, KE, 128, 1);
  };

  branch(x_m, e_mm, M,
         wx1a, F32(5),  wx1b, F32(7),  F32(8),
         wx2a, F32(10), wx2b, F32(12), F32(13),
         wlx1, F32(25), wlx2, F32(27), wlx3, F32(29), xf);

  branch(x_d, e_dd, D,
         wy1a, F32(15), wy1b, F32(17), F32(18),
         wy2a, F32(20), wy2b, F32(22), F32(23),
         wly1, F32(31), wly2, F32(33), wly3, F32(35), yf);

  // out[M,D] = xf @ yf^T : yf [D,128] is already the [N,K] operand
  gemm(xf, yf, nullptr, (float*)d_out, nullptr, M, D, KE, 0);
}
